// MultiheadAttention_21930103013844
// MI455X (gfx1250) — matrix operand order repaired and checked by reading, not compiled, not run on hardware
//
#include <hip/hip_runtime.h>

#define B_  4
#define S_  1024
#define D_  1024
#define H_  16
#define DH_ 64

typedef float    v8f  __attribute__((ext_vector_type(8)));
typedef _Float16 v16h __attribute__((ext_vector_type(16)));
typedef _Float16 v8h  __attribute__((ext_vector_type(8)));

union HFrag { v16h v; v8h h[2]; };

// ---------------------------------------------------------------- f32 -> f16
__global__ __launch_bounds__(256)
void cvt_f32_f16(const float* __restrict__ s, _Float16* __restrict__ d, int n) {
  int i = blockIdx.x * 256 + threadIdx.x;
  if (i < n) d[i] = (_Float16)s[i];
}

// ------------------------------------------- V[b,s,h,dh] -> Vt[b,h,dh,s]
__global__ __launch_bounds__(256)
void transpose_v(const _Float16* __restrict__ Vp, _Float16* __restrict__ Vt) {
  int i  = blockIdx.x * 256 + threadIdx.x;   // over B*H*DH*S = 4M
  int kk = i & (S_ - 1);
  int R  = i >> 10;          // R = b*1024 + (h*64 + d)
  int b  = R >> 10;
  int col = R & 1023;
  Vt[i] = Vp[(size_t)b * S_ * D_ + (size_t)kk * D_ + col];
}

// --------------------------------------------------------------- NT GEMM
// C[m][n] = sum_k A[m][k] * Bw[n][k]   (A: MxK row-major, Bw: NxK row-major)
// LDS tiles staged with gfx1250 GLOBAL_LOAD_ASYNC_TO_LDS_B128 (ASYNCcnt).
template<bool F32OUT>
__global__ __launch_bounds__(256)
void wmma_gemm_nt(const _Float16* __restrict__ A, const _Float16* __restrict__ Bw,
                  _Float16* __restrict__ Ch, float* __restrict__ Cf,
                  int M, int N, int K) {
  constexpr int AST = 48, BST = 48;          // half strides, 96B (16B aligned)
  __shared__ _Float16 As[128 * AST];
  __shared__ _Float16 Bs[64 * BST];
  const int tid  = threadIdx.x;
  const int lane = tid & 31;
  const int w    = tid >> 5;                 // wave 0..7 -> 16-row M strip
  const int bm   = blockIdx.x * 128;
  const int bn   = blockIdx.y * 64;
  const int ml   = lane & 15;
  const int hi   = lane >> 4;

  v8f acc[4] = {};

  for (int k0 = 0; k0 < K; k0 += 32) {
    // async cooperative loads: A 128x32, B 64x32; 16B memory->LDS per lane,
    // no VGPR staging, tracked with ASYNCcnt.
    #pragma unroll
    for (int c = 0; c < 2; ++c) {
      int idx = tid + c * 256;               // 0..511
      int r = idx >> 2, cc = (idx & 3) * 8;
      unsigned lds = (unsigned)(size_t)(void*)&As[r * AST + cc];
      const _Float16* g = &A[(size_t)(bm + r) * K + k0 + cc];
      asm volatile("global_load_async_to_lds_b128 %0, %1, off"
                   :: "v"(lds), "v"(g) : "memory");
    }
    {
      int r = tid >> 2, cc = (tid & 3) * 8;
      unsigned lds = (unsigned)(size_t)(void*)&Bs[r * BST + cc];
      const _Float16* g = &Bw[(size_t)(bn + r) * K + k0 + cc];
      asm volatile("global_load_async_to_lds_b128 %0, %1, off"
                   :: "v"(lds), "v"(g) : "memory");
    }
    if (k0 + 32 < K) {                       // hint next k-slab
      __builtin_prefetch(&A[(size_t)(bm + (tid >> 2)) * K + k0 + 32 + (tid & 3) * 8], 0, 1);
      __builtin_prefetch(&Bw[(size_t)(bn + ((tid >> 2) & 63)) * K + k0 + 32 + (tid & 3) * 8], 0, 1);
    }
    asm volatile("s_wait_asynccnt 0" ::: "memory");
    __syncthreads();

    HFrag a;                                 // A 16x32: k = {8*hi..}, {16+8*hi..}
    a.h[0] = *(const v8h*)&As[(w * 16 + ml) * AST + 8 * hi];
    a.h[1] = *(const v8h*)&As[(w * 16 + ml) * AST + 16 + 8 * hi];
    #pragma unroll
    for (int t = 0; t < 4; ++t) {            // B 32x16: k = {8*hi..}, {16+8*hi..}
      HFrag bfr;
      bfr.h[0] = *(const v8h*)&Bs[(t * 16 + ml) * BST + 8 * hi];
      bfr.h[1] = *(const v8h*)&Bs[(t * 16 + ml) * BST + 16 + 8 * hi];
      acc[t] = __builtin_amdgcn_wmma_f32_16x16x32_f16(
          false, a.v, false, bfr.v, (short)0, acc[t], false, false);
    }
    __syncthreads();
  }

  #pragma unroll
  for (int t = 0; t < 4; ++t)
    #pragma unroll
    for (int r = 0; r < 8; ++r) {
      int m = bm + w * 16 + r + 8 * hi;
      int n = bn + t * 16 + ml;
      if constexpr (F32OUT) Cf[(size_t)m * N + n] = acc[t][r];
      else                  Ch[(size_t)m * N + n] = (_Float16)acc[t][r];
    }
}

// ------------------------------------------------------- flash attention
// grid: B*H*(S/64) blocks, 128 threads (4 waves); wave w owns query rows w*16..
__global__ __launch_bounds__(128)
void flash_attn(const _Float16* __restrict__ Qp, const _Float16* __restrict__ Kp,
                const _Float16* __restrict__ Vt, const unsigned char* __restrict__ mask,
                const float* __restrict__ gamma, _Float16* __restrict__ ctx) {
  constexpr int PST = 80;                    // 160B row stride, 16B aligned
  __shared__ _Float16 Pst[4 * 16 * PST];

  const int qb = blockIdx.x & 15;
  const int h  = (blockIdx.x >> 4) & 15;
  const int b  = blockIdx.x >> 8;
  const int tid = threadIdx.x;
  const int lane = tid & 31;
  const int w    = tid >> 5;
  const int ml   = lane & 15;
  const int hi   = lane >> 4;
  const float g  = gamma[h];

  // Q A-fragments, loaded once: rows q = qb*64 + w*16 + ml, cols h*64..h*64+63
  const _Float16* Qrow = Qp + ((size_t)(b * S_ + qb * 64 + w * 16 + ml)) * D_ + h * DH_;
  HFrag qa0, qa1;
  qa0.h[0] = *(const v8h*)&Qrow[8 * hi];
  qa0.h[1] = *(const v8h*)&Qrow[16 + 8 * hi];
  qa1.h[0] = *(const v8h*)&Qrow[32 + 8 * hi];
  qa1.h[1] = *(const v8h*)&Qrow[48 + 8 * hi];

  v8f  o[4] = {};
  float rmax[8], rsum[8] = {};
  #pragma unroll
  for (int r = 0; r < 8; ++r) rmax[r] = -__builtin_inff();

  _Float16* Pw = &Pst[w * 16 * PST];

  for (int kb = 0; kb < S_ / 64; ++kb) {
    const int kbase = kb * 64;
    if (mask[(size_t)b * S_ + kbase]) break;   // monotone padding mask, uniform

    // -------- scores: Q(16x64) x K^T -> 16x64, via 2 WMMAs per 16-col tile
    v8f sc[4];
    #pragma unroll
    for (int t = 0; t < 4; ++t) {
      const _Float16* Krow =
          Kp + ((size_t)(b * S_ + kbase + t * 16 + ml)) * D_ + h * DH_;
      HFrag kb0, kb1;
      kb0.h[0] = *(const v8h*)&Krow[8 * hi];
      kb0.h[1] = *(const v8h*)&Krow[16 + 8 * hi];
      kb1.h[0] = *(const v8h*)&Krow[32 + 8 * hi];
      kb1.h[1] = *(const v8h*)&Krow[48 + 8 * hi];
      v8f c = {};
      c = __builtin_amdgcn_wmma_f32_16x16x32_f16(false, qa0.v, false, kb0.v, (short)0, c, false, false);
      c = __builtin_amdgcn_wmma_f32_16x16x32_f16(false, qa1.v, false, kb1.v, (short)0, c, false, false);
      sc[t] = c;
    }
    // scale + key padding mask (per column)
    #pragma unroll
    for (int t = 0; t < 4; ++t) {
      bool mk = mask[(size_t)b * S_ + kbase + t * 16 + ml] != 0;
      #pragma unroll
      for (int r = 0; r < 8; ++r)
        sc[t][r] = mk ? -__builtin_inff() : sc[t][r] * 0.125f;
    }
    // -------- online softmax: per-row (r) max/sum across 16 lanes of half-wave
    float bmax[8];
    #pragma unroll
    for (int r = 0; r < 8; ++r) {
      float m = sc[0][r];
      m = fmaxf(m, sc[1][r]); m = fmaxf(m, sc[2][r]); m = fmaxf(m, sc[3][r]);
      #pragma unroll
      for (int off = 1; off < 16; off <<= 1) m = fmaxf(m, __shfl_xor(m, off, 16));
      bmax[r] = m;
    }
    float alpha[8];
    #pragma unroll
    for (int r = 0; r < 8; ++r) {
      float nm = fmaxf(rmax[r], bmax[r]);
      alpha[r] = __expf(rmax[r] - nm);
      rmax[r] = nm;
    }
    float bsum[8] = {};
    #pragma unroll
    for (int t = 0; t < 4; ++t)
      #pragma unroll
      for (int r = 0; r < 8; ++r) {
        float p = __expf(sc[t][r] - rmax[r]);
        sc[t][r] = p;
        bsum[r] += p;
      }
    #pragma unroll
    for (int r = 0; r < 8; ++r) {
      float s = bsum[r];
      #pragma unroll
      for (int off = 1; off < 16; off <<= 1) s += __shfl_xor(s, off, 16);
      rsum[r] = rsum[r] * alpha[r] + s;
      #pragma unroll
      for (int t = 0; t < 4; ++t) o[t][r] *= alpha[r];
    }
    // -------- stage P (C-layout -> LDS row-major) and reload as A-fragments
    #pragma unroll
    for (int t = 0; t < 4; ++t)
      #pragma unroll
      for (int r = 0; r < 8; ++r)
        Pw[(r + 8 * hi) * PST + t * 16 + ml] = (_Float16)sc[t][r];
    __syncthreads();
    HFrag pa0, pa1;
    pa0.h[0] = *(const v8h*)&Pw[ml * PST + 8 * hi];
    pa0.h[1] = *(const v8h*)&Pw[ml * PST + 16 + 8 * hi];
    pa1.h[0] = *(const v8h*)&Pw[ml * PST + 32 + 8 * hi];
    pa1.h[1] = *(const v8h*)&Pw[ml * PST + 48 + 8 * hi];
    // -------- O += P(16x64) x V(64x64), V pre-transposed: Vt[(b,h,d)][kk]
    #pragma unroll
    for (int t = 0; t < 4; ++t) {
      const _Float16* Vrow =
          Vt + ((size_t)((b * H_ + h) * DH_ + t * 16 + ml)) * S_ + kbase;
      HFrag vb0, vb1;
      vb0.h[0] = *(const v8h*)&Vrow[8 * hi];
      vb0.h[1] = *(const v8h*)&Vrow[16 + 8 * hi];
      vb1.h[0] = *(const v8h*)&Vrow[32 + 8 * hi];
      vb1.h[1] = *(const v8h*)&Vrow[48 + 8 * hi];
      o[t] = __builtin_amdgcn_wmma_f32_16x16x32_f16(false, pa0.v, false, vb0.v, (short)0, o[t], false, false);
      o[t] = __builtin_amdgcn_wmma_f32_16x16x32_f16(false, pa1.v, false, vb1.v, (short)0, o[t], false, false);
    }
    __syncthreads();
  }

  // -------- normalize, gamma, write context (f16, [B,S,D] layout)
  float inv[8];
  #pragma unroll
  for (int r = 0; r < 8; ++r) inv[r] = g / rsum[r];
  #pragma unroll
  for (int t = 0; t < 4; ++t)
    #pragma unroll
    for (int r = 0; r < 8; ++r) {
      int q = qb * 64 + w * 16 + r + 8 * hi;
      ctx[((size_t)(b * S_ + q)) * D_ + h * DH_ + t * 16 + ml] =
          (_Float16)(o[t][r] * inv[r]);
    }
}

// ----------------------------------------------------------------- launcher
extern "C" void kernel_launch(void* const* d_in, const int* in_sizes, int n_in,
                              void* d_out, int out_size, void* d_ws, size_t ws_size,
                              hipStream_t stream) {
  const float* query = (const float*)d_in[0];
  const float* key   = (const float*)d_in[1];
  const float* value = (const float*)d_in[2];
  const unsigned char* mask = (const unsigned char*)d_in[3]; // bool (B,S)
  const float* Wq = (const float*)d_in[4];
  const float* Wk = (const float*)d_in[5];
  const float* Wv = (const float*)d_in[6];
  const float* Wo = (const float*)d_in[7];
  const float* gamma = (const float*)d_in[8];
  float* out = (float*)d_out;

  char* ws = (char*)d_ws;
  const size_t MB = 1024 * 1024;
  _Float16* qh  = (_Float16*)(ws + 0 * MB);
  _Float16* kh  = (_Float16*)(ws + 8 * MB);
  _Float16* vh  = (_Float16*)(ws + 16 * MB);
  _Float16* wqh = (_Float16*)(ws + 24 * MB);
  _Float16* wkh = (_Float16*)(ws + 26 * MB);
  _Float16* wvh = (_Float16*)(ws + 28 * MB);
  _Float16* woh = (_Float16*)(ws + 30 * MB);
  _Float16* Qp  = (_Float16*)(ws + 32 * MB);
  _Float16* Kp  = (_Float16*)(ws + 40 * MB);
  _Float16* Vp  = (_Float16*)(ws + 48 * MB);
  _Float16* Vt  = (_Float16*)(ws + 56 * MB);
  _Float16* ctx = (_Float16*)(ws + 64 * MB);   // 72 MB total

  const int MT = B_ * S_;                       // 4096 rows

  cvt_f32_f16<<<MT * D_ / 256, 256, 0, stream>>>(query, qh, MT * D_);
  cvt_f32_f16<<<MT * D_ / 256, 256, 0, stream>>>(key,   kh, MT * D_);
  cvt_f32_f16<<<MT * D_ / 256, 256, 0, stream>>>(value, vh, MT * D_);
  cvt_f32_f16<<<D_ * D_ / 256, 256, 0, stream>>>(Wq, wqh, D_ * D_);
  cvt_f32_f16<<<D_ * D_ / 256, 256, 0, stream>>>(Wk, wkh, D_ * D_);
  cvt_f32_f16<<<D_ * D_ / 256, 256, 0, stream>>>(Wv, wvh, D_ * D_);
  cvt_f32_f16<<<D_ * D_ / 256, 256, 0, stream>>>(Wo, woh, D_ * D_);

  dim3 ggrid(MT / 128, D_ / 64);
  wmma_gemm_nt<false><<<ggrid, 256, 0, stream>>>(qh, wqh, Qp, nullptr, MT, D_, D_);
  wmma_gemm_nt<false><<<ggrid, 256, 0, stream>>>(kh, wkh, Kp, nullptr, MT, D_, D_);
  wmma_gemm_nt<false><<<ggrid, 256, 0, stream>>>(vh, wvh, Vp, nullptr, MT, D_, D_);

  transpose_v<<<(B_ * H_ * DH_ * S_) / 256, 256, 0, stream>>>(Vp, Vt);

  flash_attn<<<B_ * H_ * (S_ / 64), 128, 0, stream>>>(Qp, Kp, Vt, mask, gamma, ctx);

  wmma_gemm_nt<true><<<ggrid, 256, 0, stream>>>(ctx, woh, nullptr, out, MT, D_, D_);
}
